// VAEHyperNetFusion_57758720197281
// MI455X (gfx1250) — compile-verified, matrix operand order checked by reading, not run on hardware
//
#include <hip/hip_runtime.h>
#include <hip/hip_bf16.h>

typedef __attribute__((ext_vector_type(16))) _Float16 v16h;
typedef __attribute__((ext_vector_type(8)))  _Float16 v8h;
typedef __attribute__((ext_vector_type(8)))  float    v8f;
typedef __attribute__((ext_vector_type(4)))  float    v4f;

#define TRAIN_N 8192
#define TEST_N  32768
#define DD      512
#define TT      64
#define NI      7
#define NLEAF   8
#define PPT     3607      // NI*DD + NI + NLEAF*2
#define TOTP    230912    // PPT*TT + TT
#define HID     64
#define HID2    128
#define NROWS   448       // TT*NI

static __device__ __forceinline__ v16h pack16(v8h lo, v8h hi) {
  v16h r;
#pragma unroll
  for (int i = 0; i < 8; ++i) { r[i] = lo[i]; r[i + 8] = hi[i]; }
  return r;
}

static __device__ __forceinline__ v8f wmma_f16(v16h a, v16h b, v8f c) {
  return __builtin_amdgcn_wmma_f32_16x16x32_f16(false, a, false, b, (short)0, c,
                                                false, false);
}

// ---------------- K0: convert W1,W2 to f16; zero S ----------------
__global__ void k0_prep(const float* __restrict__ W1, const float* __restrict__ W2,
                        _Float16* __restrict__ W1h, _Float16* __restrict__ W2h,
                        float* __restrict__ S) {
  int n = blockIdx.x * 256 + threadIdx.x;
  if (n < HID * DD) {
    W1h[n] = (_Float16)W1[n];
  } else if (n < HID * DD + HID * HID) {
    int m = n - HID * DD;
    W2h[m] = (_Float16)W2[m];
  } else if (n < HID * DD + HID * HID + HID) {
    S[n - (HID * DD + HID * HID)] = 0.0f;
  }
}

// ---------------- K1: trunk GEMMs + LN + column-sum ----------------
// grid 128 blocks x 128 threads; each block: 64 rows of X_train
__global__ void __launch_bounds__(128)
k1_trunk(const float* __restrict__ X, const float* __restrict__ b1,
         const float* __restrict__ g1, const float* __restrict__ be1,
         const _Float16* __restrict__ W1h, const _Float16* __restrict__ W2h,
         const float* __restrict__ b2, float* __restrict__ S) {
  __shared__ float hbuf[64][68];                       // pre-LN activations
  __shared__ __align__(16) _Float16 h1buf[64][72];     // post-LN f16 (WMMA A)
  __shared__ float colsum[64];

  const int tid = threadIdx.x;
  const int lane = tid & 31, w = tid >> 5;
  const int lmod = lane & 15, lhi = lane >> 4;
  const int rows0 = blockIdx.x * 64;

  // ---- GEMM1: X(64x512) @ W1^T(512x64) ----
  const int m = w * 16 + lmod;
  const float* xrow = X + (size_t)(rows0 + m) * DD;
  v8f acc[4];
#pragma unroll
  for (int nt = 0; nt < 4; ++nt) acc[nt] = (v8f){0,0,0,0,0,0,0,0};

  for (int ks = 0; ks < DD / 32; ++ks) {
    const int k0 = ks * 32 + lhi * 8;
    float4 f0 = *(const float4*)(xrow + k0);
    float4 f1 = *(const float4*)(xrow + k0 + 4);
    float4 f2 = *(const float4*)(xrow + k0 + 16);
    float4 f3 = *(const float4*)(xrow + k0 + 20);
    v16h a;
    a[0]=(_Float16)f0.x; a[1]=(_Float16)f0.y; a[2]=(_Float16)f0.z; a[3]=(_Float16)f0.w;
    a[4]=(_Float16)f1.x; a[5]=(_Float16)f1.y; a[6]=(_Float16)f1.z; a[7]=(_Float16)f1.w;
    a[8]=(_Float16)f2.x; a[9]=(_Float16)f2.y; a[10]=(_Float16)f2.z; a[11]=(_Float16)f2.w;
    a[12]=(_Float16)f3.x; a[13]=(_Float16)f3.y; a[14]=(_Float16)f3.z; a[15]=(_Float16)f3.w;
#pragma unroll
    for (int nt = 0; nt < 4; ++nt) {
      const _Float16* wr = W1h + (size_t)(nt * 16 + lmod) * DD;
      v8h blo = *(const v8h*)(wr + k0);
      v8h bhi = *(const v8h*)(wr + k0 + 16);
      acc[nt] = wmma_f16(a, pack16(blo, bhi), acc[nt]);
    }
  }
  // spill +b1 to LDS for LayerNorm
#pragma unroll
  for (int nt = 0; nt < 4; ++nt) {
    const int n = nt * 16 + lmod;
#pragma unroll
    for (int j = 0; j < 8; ++j)
      hbuf[w * 16 + j + lhi * 8][n] = acc[nt][j] + b1[n];
  }
  __syncthreads();

  // ---- LayerNorm + relu -> f16 ----
  if (tid < 64) {
    float mean = 0.f;
#pragma unroll 8
    for (int n = 0; n < 64; ++n) mean += hbuf[tid][n];
    mean *= (1.0f / 64.0f);
    float var = 0.f;
#pragma unroll 8
    for (int n = 0; n < 64; ++n) { float d = hbuf[tid][n] - mean; var += d * d; }
    var *= (1.0f / 64.0f);
    const float sc = rsqrtf(var + 1e-5f);
#pragma unroll 8
    for (int n = 0; n < 64; ++n) {
      float v = (hbuf[tid][n] - mean) * sc * g1[n] + be1[n];
      h1buf[tid][n] = (_Float16)fmaxf(v, 0.0f);
    }
    colsum[tid] = 0.0f;
  }
  __syncthreads();

  // ---- GEMM2: h1(64x64) @ W2^T(64x64), relu, column sums ----
  v8f acc2[4];
#pragma unroll
  for (int nt = 0; nt < 4; ++nt) acc2[nt] = (v8f){0,0,0,0,0,0,0,0};
#pragma unroll
  for (int ks = 0; ks < 2; ++ks) {
    const int k0 = ks * 32 + lhi * 8;
    v8h alo = *(const v8h*)&h1buf[w * 16 + lmod][k0];
    v8h ahi = *(const v8h*)&h1buf[w * 16 + lmod][k0 + 16];
    v16h a = pack16(alo, ahi);
#pragma unroll
    for (int nt = 0; nt < 4; ++nt) {
      const _Float16* wr = W2h + (size_t)(nt * 16 + lmod) * HID;
      v8h blo = *(const v8h*)(wr + k0);
      v8h bhi = *(const v8h*)(wr + k0 + 16);
      acc2[nt] = wmma_f16(a, pack16(blo, bhi), acc2[nt]);
    }
  }
#pragma unroll
  for (int nt = 0; nt < 4; ++nt) {
    const int n = nt * 16 + lmod;
    const float bb = b2[n];
    float s = 0.f;
#pragma unroll
    for (int j = 0; j < 8; ++j) s += fmaxf(acc2[nt][j] + bb, 0.0f);
    atomicAdd(&colsum[n], s);
  }
  __syncthreads();
  if (tid < 64) atomicAdd(&S[tid], colsum[tid]);
}

// ---------------- K2: hypernet head (single block, 128 threads) ----------------
__global__ void k2_head(const float* __restrict__ S, const float* __restrict__ H1,
                        const float* __restrict__ hb1, const float* __restrict__ g2,
                        const float* __restrict__ be2, const float* __restrict__ H2,
                        const float* __restrict__ hb2, float* __restrict__ u2) {
  __shared__ float cs[64];
  __shared__ float u1s[128];
  __shared__ float red[128];
  const int t = threadIdx.x;
  if (t < 64) cs[t] = S[t] * (1.0f / (float)TRAIN_N);
  __syncthreads();

  float a = hb1[t];
#pragma unroll 8
  for (int k = 0; k < 64; ++k) a += H1[t * 64 + k] * cs[k];

  red[t] = a; __syncthreads();
  for (int s = 64; s > 0; s >>= 1) { if (t < s) red[t] += red[t + s]; __syncthreads(); }
  const float mean = red[0] * (1.0f / 128.0f);
  __syncthreads();
  const float d = a - mean;
  red[t] = d * d; __syncthreads();
  for (int s = 64; s > 0; s >>= 1) { if (t < s) red[t] += red[t + s]; __syncthreads(); }
  const float var = red[0] * (1.0f / 128.0f);
  const float sc = rsqrtf(var + 1e-5f);
  u1s[t] = fmaxf(d * sc * g2[t] + be2[t], 0.0f);
  __syncthreads();

  float b = hb2[t];
#pragma unroll 8
  for (int k = 0; k < 128; ++k) b += H2[t * 128 + k] * u1s[k];
  u2[t] = fmaxf(b, 0.0f);
}

// ---------------- K3: params = u2 @ H3^T + hb3, scatter ----------------
__global__ void __launch_bounds__(256)
k3_params(const float* __restrict__ H3, const float* __restrict__ hb3,
          const float* __restrict__ u2, _Float16* __restrict__ swh,
          float* __restrict__ sb, float* __restrict__ llw,
          float* __restrict__ tww) {
  __shared__ __align__(16) float u2s[128];
  const int tid = threadIdx.x;
  if (tid < 128) u2s[tid] = u2[tid];
  __syncthreads();

  const int r = blockIdx.x * 256 + tid;
  if (r >= TOTP) return;

  const v4f* hv = (const v4f*)(H3 + (size_t)r * HID2);
  const v4f* uv = (const v4f*)u2s;
  float dot = 0.f;
#pragma unroll 8
  for (int i = 0; i < 32; ++i) {
    v4f h = __builtin_nontemporal_load(hv + i);  // H3 is single-use: keep L2 clean
    v4f u = uv[i];
    dot += h[0] * u[0] + h[1] * u[1] + h[2] * u[2] + h[3] * u[3];
  }
  const float val = dot + hb3[r];

  if (r < TT * PPT) {
    const int t = r / PPT;
    const int j = r - t * PPT;
    if (j < NI * DD)            swh[(size_t)t * (NI * DD) + j] = (_Float16)val;
    else if (j < NI * DD + NI)  sb[t * NI + (j - NI * DD)] = val;
    else                        llw[t * 16 + (j - NI * DD - NI)] = val;
  } else {
    tww[r - TT * PPT] = val;
  }
}

// ---------------- K4: fold softmax(tw), softmax(ll), x2 into L ----------------
__global__ void k4_fold(const float* __restrict__ tww, const float* __restrict__ llw,
                        float* __restrict__ L) {
  __shared__ float red[64];
  const int t = threadIdx.x;
  const float tv = tww[t];
  red[t] = tv; __syncthreads();
  for (int s = 32; s > 0; s >>= 1) { if (t < s) red[t] = fmaxf(red[t], red[t + s]); __syncthreads(); }
  const float mx = red[0];
  __syncthreads();
  const float e = __expf(tv - mx);
  red[t] = e; __syncthreads();
  for (int s = 32; s > 0; s >>= 1) { if (t < s) red[t] += red[t + s]; __syncthreads(); }
  const float w = e / red[0];

#pragma unroll
  for (int l = 0; l < NLEAF; ++l) {
    const float a0 = llw[t * 16 + l * 2];
    const float a1 = llw[t * 16 + l * 2 + 1];
    const float m = fmaxf(a0, a1);
    const float e0 = __expf(a0 - m), e1 = __expf(a1 - m);
    const float s = 2.0f * w / (e0 + e1);
    L[t * 16 + l * 2]     = e0 * s;
    L[t * 16 + l * 2 + 1] = e1 * s;
  }
}

// ---------------- K5: fused big GEMM + sigmoid + tree routing ----------------
// grid 1024 x 128 threads; each block: 32 rows of X_test
// K-outer register blocking: 2 M-subtiles x 7 N-tiles = 14 live accumulators;
// per K-step: 4 ds_load_b128 (A) + 14 global_load_b128 (B) -> 14 WMMAs.
__global__ void __launch_bounds__(128)
k5_forest(const float* __restrict__ Xt, const _Float16* __restrict__ swh,
          const float* __restrict__ sb, const float* __restrict__ L,
          float* __restrict__ out) {
  __shared__ __align__(16) _Float16 xls[32][520];   // padded: bank-friendly, 16B rows
  __shared__ _Float16 pls[32][NROWS];
  __shared__ float outacc[32][2];

  const int tid = threadIdx.x;
  const int lane = tid & 31, w = tid >> 5;
  const int lmod = lane & 15, lhi = lane >> 4;
  const int rows0 = blockIdx.x * 32;

  // stage X tile -> f16 LDS (32x512), 4 elems per iter
  for (int i = tid; i < (32 * DD) / 4; i += 128) {
    const int r = (i * 4) / DD, c = (i * 4) % DD;
    float4 v = *(const float4*)(Xt + (size_t)(rows0 + r) * DD + c);
    xls[r][c]     = (_Float16)v.x;
    xls[r][c + 1] = (_Float16)v.y;
    xls[r][c + 2] = (_Float16)v.z;
    xls[r][c + 3] = (_Float16)v.w;
  }
  if (tid < 64) outacc[tid >> 1][tid & 1] = 0.0f;
  __syncthreads();

  // ---- GEMM: (32x512) @ swh^T(512x448), wave w owns N-tiles [7w, 7w+7) ----
  v8f acc[2][7];
#pragma unroll
  for (int ms = 0; ms < 2; ++ms)
#pragma unroll
    for (int nti = 0; nti < 7; ++nti) acc[ms][nti] = (v8f){0,0,0,0,0,0,0,0};

  // per-lane base pointer into swh for this wave's N-tile group
  const _Float16* wbase = swh + (size_t)(w * 7 * 16 + lmod) * DD;

#pragma unroll 2
  for (int ks = 0; ks < DD / 32; ++ks) {
    const int k0 = ks * 32 + lhi * 8;
    v16h a0 = pack16(*(const v8h*)&xls[lmod][k0],
                     *(const v8h*)&xls[lmod][k0 + 16]);
    v16h a1 = pack16(*(const v8h*)&xls[16 + lmod][k0],
                     *(const v8h*)&xls[16 + lmod][k0 + 16]);
#pragma unroll
    for (int nti = 0; nti < 7; ++nti) {
      const _Float16* wr = wbase + (size_t)(nti * 16) * DD;
      v16h b = pack16(*(const v8h*)(wr + k0), *(const v8h*)(wr + k0 + 16));
      acc[0][nti] = wmma_f16(a0, b, acc[0][nti]);
      acc[1][nti] = wmma_f16(a1, b, acc[1][nti]);
    }
  }

  // epilogue: bias + sigmoid -> pls
#pragma unroll
  for (int ms = 0; ms < 2; ++ms) {
#pragma unroll
    for (int nti = 0; nti < 7; ++nti) {
      const int n = (w * 7 + nti) * 16 + lmod;
      const float bias = sb[n];
#pragma unroll
      for (int j = 0; j < 8; ++j) {
        const float x = acc[ms][nti][j] + bias;
        const float p = 1.0f / (1.0f + __expf(-x));
        pls[ms * 16 + j + lhi * 8][n] = (_Float16)p;
      }
    }
  }
  __syncthreads();

  // ---- soft-tree routing: thread -> (row = tid&31, trees tg*16..+15) ----
  {
    const int r = tid & 31;
    const int tg = tid >> 5;
    float o0 = 0.f, o1 = 0.f;
    for (int ti = 0; ti < 16; ++ti) {
      const int t = tg * 16 + ti;
      float p[NI];
#pragma unroll
      for (int i = 0; i < NI; ++i) p[i] = (float)pls[r][t * NI + i];
      const float q0 = 1.f - p[0];
      float lf[NLEAF];
      lf[0] = q0   * (1.f - p[1]) * (1.f - p[3]);
      lf[1] = q0   * (1.f - p[1]) * p[3];
      lf[2] = q0   * p[1]         * (1.f - p[4]);
      lf[3] = q0   * p[1]         * p[4];
      lf[4] = p[0] * (1.f - p[2]) * (1.f - p[5]);
      lf[5] = p[0] * (1.f - p[2]) * p[5];
      lf[6] = p[0] * p[2]         * (1.f - p[6]);
      lf[7] = p[0] * p[2]         * p[6];
      const float* Lr = L + t * 16;
#pragma unroll
      for (int l = 0; l < NLEAF; ++l) {
        o0 += lf[l] * Lr[l * 2];
        o1 += lf[l] * Lr[l * 2 + 1];
      }
    }
    atomicAdd(&outacc[r][0], o0);
    atomicAdd(&outacc[r][1], o1);
  }
  __syncthreads();
  if (tid < 64)
    out[(size_t)(rows0 + (tid >> 1)) * 2 + (tid & 1)] = outacc[tid >> 1][tid & 1];
}

// ---------------- host launcher ----------------
extern "C" void kernel_launch(void* const* d_in, const int* in_sizes, int n_in,
                              void* d_out, int out_size, void* d_ws, size_t ws_size,
                              hipStream_t stream) {
  const float* X_train = (const float*)d_in[0];
  const float* X_test  = (const float*)d_in[1];
  // d_in[2] = y_train (unused by reference forward path)
  const float* W1  = (const float*)d_in[3];
  const float* b1  = (const float*)d_in[4];
  const float* g1  = (const float*)d_in[5];
  const float* be1 = (const float*)d_in[6];
  const float* W2  = (const float*)d_in[7];
  const float* b2  = (const float*)d_in[8];
  const float* H1  = (const float*)d_in[9];
  const float* hb1 = (const float*)d_in[10];
  const float* g2  = (const float*)d_in[11];
  const float* be2 = (const float*)d_in[12];
  const float* H2  = (const float*)d_in[13];
  const float* hb2 = (const float*)d_in[14];
  const float* H3  = (const float*)d_in[15];
  const float* hb3 = (const float*)d_in[16];
  float* out = (float*)d_out;

  // workspace carve-up (bytes)
  char* ws = (char*)d_ws;
  _Float16* W1h = (_Float16*)(ws);                 // 65536
  _Float16* W2h = (_Float16*)(ws + 65536);         // 8192
  float*    S   = (float*)(ws + 73728);            // 256
  float*    u2  = (float*)(ws + 73984);            // 512
  _Float16* swh = (_Float16*)(ws + 74496);         // 458752
  float*    sb  = (float*)(ws + 533248);           // 1792
  float*    llw = (float*)(ws + 535040);           // 4096
  float*    tww = (float*)(ws + 539136);           // 256
  float*    L   = (float*)(ws + 539392);           // 4096
  if (ws_size < 543488) return;

  k0_prep<<<(HID * DD + HID * HID + HID + 255) / 256, 256, 0, stream>>>(
      W1, W2, W1h, W2h, S);
  k1_trunk<<<TRAIN_N / 64, 128, 0, stream>>>(X_train, b1, g1, be1, W1h, W2h, b2, S);
  k2_head<<<1, 128, 0, stream>>>(S, H1, hb1, g2, be2, H2, hb2, u2);
  k3_params<<<(TOTP + 255) / 256, 256, 0, stream>>>(H3, hb3, u2, swh, sb, llw, tww);
  k4_fold<<<1, 64, 0, stream>>>(tww, llw, L);
  k5_forest<<<TEST_N / 32, 128, 0, stream>>>(X_test, swh, sb, L, out);
}
